// MultiHeadAttention_7876970021382
// MI455X (gfx1250) — compile-verified, matrix operand order checked by reading, not run on hardware
//
#include <hip/hip_runtime.h>
#include <hip/hip_bf16.h>

// ---------------------------------------------------------------------------
// MHA forward for MI455X (gfx1250, wave32, WMMA).
//   B=2, S=2048, D=1024, H=16, dh=64.
// f16 data / f32 accumulate via v_wmma_f32_16x16x32_f16.
//   1) cast x, Wq, Wk, Wv, Wo -> f16 workspace
//   2) WMMA GEMM (LDS-staged, async-to-LDS double buffered, block 128x128,
//      wave 32x64): Q,K -> [b,h,s,dh] f16 ; V -> [b,h,dh,s] f16
//   3) flash-attention, 1 wave per (b,h,16 query rows), online softmax,
//      P staged through LDS to re-layout D-frag -> A-frag
//   4) same GEMM: out = attn @ Wo^T + bo (f32 out)
// ---------------------------------------------------------------------------

typedef _Float16 v8h  __attribute__((ext_vector_type(8)));
typedef _Float16 v16h __attribute__((ext_vector_type(16)));
typedef float    v8f  __attribute__((ext_vector_type(8)));
typedef int      v4i  __attribute__((vector_size(16)));   // matches builtin sig

#define D_MODEL 1024
#define NUM_HEADS 16
#define D_HEAD 64
#define BATCH 2
#define SEQ 2048
#define MROWS (BATCH * SEQ)           // 4096

// gfx1250 async global->LDS path (ASYNCcnt-tracked), with portable fallback.
#if defined(__gfx1250__) && __has_builtin(__builtin_amdgcn_global_load_async_to_lds_b128)
#define HAS_ASYNC_LDS 1
#else
#define HAS_ASYNC_LDS 0
#endif

#if HAS_ASYNC_LDS
typedef __attribute__((address_space(1))) v4i* g_v4i_p;   // global (device) AS
typedef __attribute__((address_space(3))) v4i* l_v4i_p;   // LDS (shared) AS
#endif

__device__ __forceinline__ v8f wmma_f16(v16h a, v16h b, v8f c) {
  return __builtin_amdgcn_wmma_f32_16x16x32_f16(
      /*neg_a=*/false, a, /*neg_b=*/false, b,
      /*c_mod=*/(short)0, c, /*reuse_a=*/false, /*reuse_b=*/false);
}

// 16x32 f16 A-fragment (or B^T-fragment) from a row-major tile:
//  lane L: row = L&15, halves 0..7 = K koff..koff+7, halves 8..15 = 16+koff..,
//  koff = 8*(L>>4).   (CDNA5 ISA 7.12.2)
__device__ __forceinline__ v16h load_frag(const _Float16* base, long ld) {
  const int lane = threadIdx.x & 31;
  const int r    = lane & 15;
  const int koff = (lane >> 4) * 8;
  const _Float16* p = base + (long)r * ld + koff;
  v8h lo = *(const v8h*)(p);
  v8h hi = *(const v8h*)(p + 16);
  v16h f;
#pragma unroll
  for (int i = 0; i < 8; ++i) { f[i] = lo[i]; f[8 + i] = hi[i]; }
  return f;
}

// Same fragment layout, but from an LDS tile with fixed ld = 32 halves.
__device__ __forceinline__ v16h lds_frag(const _Float16* base) {
  const int lane = threadIdx.x & 31;
  const int r    = lane & 15;
  const int koff = (lane >> 4) * 8;
  const _Float16* p = base + r * 32 + koff;
  v8h lo = *(const v8h*)(p);
  v8h hi = *(const v8h*)(p + 16);
  v16h f;
#pragma unroll
  for (int i = 0; i < 8; ++i) { f[i] = lo[i]; f[8 + i] = hi[i]; }
  return f;
}

// Stage a 128x32 f16 tile (row-major, ld elements) into LDS. 256 threads,
// 2 x 16B chunks per thread. Async (ASYNCcnt) when available.
__device__ __forceinline__ void stage_tile(_Float16* lds_dst, const _Float16* gsrc,
                                           long ld, int tid) {
#pragma unroll
  for (int c = 0; c < 2; ++c) {
    const int id  = tid + c * 256;
    const int row = id >> 2;
    const int col = (id & 3) * 8;
    const _Float16* gp = gsrc + (long)row * ld + col;
    _Float16* lp = lds_dst + row * 32 + col;
#if HAS_ASYNC_LDS
    __builtin_amdgcn_global_load_async_to_lds_b128((g_v4i_p)gp, (l_v4i_p)lp, 0, 0);
#else
    *(v8h*)lp = *(const v8h*)gp;
#endif
  }
}

template <int N>
__device__ __forceinline__ void wait_async_le() {
#if HAS_ASYNC_LDS
#if __has_builtin(__builtin_amdgcn_s_wait_asynccnt)
  __builtin_amdgcn_s_wait_asynccnt(N);
#else
  asm volatile("s_wait_asynccnt %0" :: "i"(N) : "memory");
#endif
#endif
}

// ---------------------------------------------------------------------------
// f32 -> f16 cast
// ---------------------------------------------------------------------------
__global__ void mha_cast_f16(const float* __restrict__ in,
                             _Float16* __restrict__ out, long n) {
  long i = (long)blockIdx.x * blockDim.x + threadIdx.x;
  long stride = (long)gridDim.x * blockDim.x;
  for (; i < n; i += stride) out[i] = (_Float16)in[i];
}

// ---------------------------------------------------------------------------
// WMMA GEMM: C[M,N] = A[M,K](f16) @ Bw[N,K](f16)^T + bias[N]
//  Block tile 128x128, K-chunk 32, double-buffered async LDS staging.
//  8 waves; wave tile 32x64 (2 m-frags x 4 n-frags, 8 WMMA / chunk).
//  mode 0: f32 row-major out       (O projection)
//  mode 1: f16 out at [b,h,s,dh]   (Q, K)
//  mode 2: f16 out at [b,h,dh,s]   (V transposed)
// ---------------------------------------------------------------------------
__global__ __launch_bounds__(256) void mha_gemm_wmma(
    const _Float16* __restrict__ A, const _Float16* __restrict__ Bw,
    const float* __restrict__ bias, float* __restrict__ outF,
    _Float16* __restrict__ outH, int K, int N, int mode) {
  __shared__ __align__(16) _Float16 As[2][128 * 32];   // 8 KB x2
  __shared__ __align__(16) _Float16 Bs[2][128 * 32];   // 8 KB x2

  const int tid  = threadIdx.x;
  const int wid  = tid >> 5;
  const int lane = tid & 31;
  const int gm0 = blockIdx.y * 128;
  const int gn0 = blockIdx.x * 128;
  const int wm  = (wid >> 1) * 32;   // wave row origin in block tile
  const int wn  = (wid & 1) * 64;    // wave col origin in block tile

  v8f acc[2][4] = {};

  // prologue: stage chunk 0 (4 async ops / thread)
  stage_tile(As[0], A  + (long)gm0 * K, K, tid);
  stage_tile(Bs[0], Bw + (long)gn0 * K, K, tid);

  const int nchunks = K / 32;
  for (int c = 0; c < nchunks; ++c) {
    const int buf = c & 1;
    if (c + 1 < nchunks) {  // prefetch next chunk into the other buffer
      stage_tile(As[buf ^ 1], A  + (long)gm0 * K + (c + 1) * 32, K, tid);
      stage_tile(Bs[buf ^ 1], Bw + (long)gn0 * K + (c + 1) * 32, K, tid);
      wait_async_le<4>();   // in-order completion => chunk c is resident
    } else {
      wait_async_le<0>();
    }
    __syncthreads();        // chunk c visible to all waves

    v16h a0 = lds_frag(&As[buf][(wm + 0)  * 32]);
    v16h a1 = lds_frag(&As[buf][(wm + 16) * 32]);
#pragma unroll
    for (int ns = 0; ns < 4; ++ns) {
      v16h bf = lds_frag(&Bs[buf][(wn + ns * 16) * 32]);
      acc[0][ns] = wmma_f16(a0, bf, acc[0][ns]);
      acc[1][ns] = wmma_f16(a1, bf, acc[1][ns]);
    }
    __syncthreads();        // all reads of buf done before it is overwritten
  }

  const int r  = lane & 15;
  const int rh = lane >> 4;
#pragma unroll
  for (int ms = 0; ms < 2; ++ms) {
#pragma unroll
    for (int ns = 0; ns < 4; ++ns) {
      const int n = gn0 + wn + ns * 16 + r;
      const float bn = bias[n];
#pragma unroll
      for (int j = 0; j < 8; ++j) {
        const int m = gm0 + wm + ms * 16 + rh * 8 + j;
        const float v = acc[ms][ns][j] + bn;
        if (mode == 0) {
          outF[(long)m * N + n] = v;
        } else {
          const int b = m >> 11, s = m & 2047;   // m over B*S
          const int h = n >> 6,  d = n & 63;     // n over H*dh
          if (mode == 1)
            outH[(((long)(b * NUM_HEADS + h) * SEQ) + s) * D_HEAD + d] = (_Float16)v;
          else
            outH[(((long)(b * NUM_HEADS + h) * D_HEAD) + d) * SEQ + s] = (_Float16)v;
        }
      }
    }
  }
}

// ---------------------------------------------------------------------------
// Flash attention. One wave (32 threads) per (b, h, 16 query rows).
//  Q,K: [b,h,s,64] f16 ; V: [b,h,64,s] f16 (transposed) ; out: [b,s,h*64] f16
// ---------------------------------------------------------------------------
__global__ __launch_bounds__(32) void mha_flash_attn(
    const _Float16* __restrict__ Qh, const _Float16* __restrict__ Kh,
    const _Float16* __restrict__ Vt, const int* __restrict__ pmask,
    _Float16* __restrict__ attnOut) {
  const int lane = threadIdx.x & 31;
  const int r    = lane & 15;
  const int rh   = lane >> 4;
  const int q0 = blockIdx.x * 16;
  const int h  = blockIdx.y;
  const int b  = blockIdx.z;

  const long headOff = (long)(b * NUM_HEADS + h) * SEQ * D_HEAD;
  const _Float16* Qb = Qh + headOff;  // [s][64]
  const _Float16* Kb = Kh + headOff;  // [s][64]
  const _Float16* Vb = Vt + headOff;  // [64][2048]

  __shared__ __align__(16) _Float16 pls[16 * 32];   // P tile staging, 1 KB

  const v16h qa0 = load_frag(Qb + (long)q0 * D_HEAD + 0,  D_HEAD);
  const v16h qa1 = load_frag(Qb + (long)q0 * D_HEAD + 32, D_HEAD);

  v8f o[4] = {};          // 16x64 f32 accumulator, 4 N-subtiles
  float rM[8], rS[8];
#pragma unroll
  for (int j = 0; j < 8; ++j) { rM[j] = -__builtin_inff(); rS[j] = 0.0f; }

  const float scale = 0.125f;  // 1/sqrt(64)
  const int ktmax = (q0 + 15) / 32;

  for (int kt = 0; kt <= ktmax; ++kt) {
    const int k0 = kt * 32;

    // ---- scores: S = Q @ K^T  (two 16x16 key subtiles) ----
    v8f s0 = {}, s1 = {};
    {
      v16h kb = load_frag(Kb + (long)k0 * D_HEAD + 0,  D_HEAD);
      s0 = wmma_f16(qa0, kb, s0);
      kb = load_frag(Kb + (long)k0 * D_HEAD + 32, D_HEAD);
      s0 = wmma_f16(qa1, kb, s0);
      kb = load_frag(Kb + (long)(k0 + 16) * D_HEAD + 0,  D_HEAD);
      s1 = wmma_f16(qa0, kb, s1);
      kb = load_frag(Kb + (long)(k0 + 16) * D_HEAD + 32, D_HEAD);
      s1 = wmma_f16(qa1, kb, s1);
    }

    const int key0 = k0 + r;
    const int key1 = k0 + 16 + r;
    const int pm0 = pmask[b * SEQ + key0];
    const int pm1 = pmask[b * SEQ + key1];

    // ---- online softmax per row (VGPR j <-> row rh*8+j) ----
#pragma unroll
    for (int j = 0; j < 8; ++j) {
      const int q = q0 + rh * 8 + j;
      float v0 = s0[j] * scale;
      float v1 = s1[j] * scale;
      if (!pm0 || key0 > q) v0 = -__builtin_inff();
      if (!pm1 || key1 > q) v1 = -__builtin_inff();

      float t = fmaxf(v0, v1);
#pragma unroll
      for (int m = 1; m < 16; m <<= 1) t = fmaxf(t, __shfl_xor(t, m, 32));
      const float nm = fmaxf(rM[j], t);
      const float corr = __expf(rM[j] - nm);
      rM[j] = nm;

      const float p0 = __expf(v0 - nm);
      const float p1 = __expf(v1 - nm);
      float ps = p0 + p1;
#pragma unroll
      for (int m = 1; m < 16; m <<= 1) ps += __shfl_xor(ps, m, 32);
      rS[j] = rS[j] * corr + ps;

      o[0][j] *= corr; o[1][j] *= corr; o[2][j] *= corr; o[3][j] *= corr;

      const int mrow = rh * 8 + j;
      pls[mrow * 32 + r]      = (_Float16)p0;
      pls[mrow * 32 + 16 + r] = (_Float16)p1;
    }

    // DS stores must land before the A-fragment DS loads below.
    asm volatile("s_wait_dscnt 0" ::: "memory");

    // ---- reload P in A-fragment layout from LDS ----
    v16h pa;
    {
      const int base = r * 32 + rh * 8;
#pragma unroll
      for (int i = 0; i < 8; ++i) {
        pa[i]     = pls[base + i];
        pa[8 + i] = pls[base + 16 + i];
      }
    }

    // ---- O += P @ V  (B-frags from V^T [d][s], contiguous rows) ----
#pragma unroll
    for (int t = 0; t < 4; ++t) {
      v16h vb = load_frag(Vb + (long)(t * 16) * SEQ + k0, SEQ);
      o[t] = wmma_f16(pa, vb, o[t]);
    }
  }

  // ---- epilogue: normalize, write [b, s, h*64+d] f16 ----
#pragma unroll
  for (int t = 0; t < 4; ++t) {
#pragma unroll
    for (int j = 0; j < 8; ++j) {
      const float v = o[t][j] / rS[j];
      const int m = rh * 8 + j;
      const int d = t * 16 + r;
      attnOut[((long)(b * SEQ + q0 + m)) * D_MODEL + h * D_HEAD + d] = (_Float16)v;
    }
  }
}

// ---------------------------------------------------------------------------
// launch
// ---------------------------------------------------------------------------
extern "C" void kernel_launch(void* const* d_in, const int* in_sizes, int n_in,
                              void* d_out, int out_size, void* d_ws, size_t ws_size,
                              hipStream_t stream) {
  (void)in_sizes; (void)n_in; (void)out_size; (void)ws_size;

  const float* x  = (const float*)d_in[0];
  const int*   pm = (const int*)d_in[1];
  const float* Wq = (const float*)d_in[2];
  const float* bq = (const float*)d_in[3];
  const float* Wk = (const float*)d_in[4];
  const float* bk = (const float*)d_in[5];
  const float* Wv = (const float*)d_in[6];
  const float* bv = (const float*)d_in[7];
  const float* Wo = (const float*)d_in[8];
  const float* bo = (const float*)d_in[9];
  float* out = (float*)d_out;

  char* ws = (char*)d_ws;
  const long NX = (long)MROWS * D_MODEL;     // 4M
  const long NW = (long)D_MODEL * D_MODEL;   // 1M
  _Float16* xh  = (_Float16*)(ws);                          // 8 MiB
  _Float16* wqh = (_Float16*)(ws + (8l << 20));             // 2 MiB each
  _Float16* wkh = (_Float16*)(ws + (10l << 20));
  _Float16* wvh = (_Float16*)(ws + (12l << 20));
  _Float16* woh = (_Float16*)(ws + (14l << 20));
  _Float16* qh  = (_Float16*)(ws + (16l << 20));            // 8 MiB each
  _Float16* kh  = (_Float16*)(ws + (24l << 20));
  _Float16* vt  = (_Float16*)(ws + (32l << 20));
  _Float16* ah  = (_Float16*)(ws + (40l << 20));

  // 1) casts
  mha_cast_f16<<<2048, 256, 0, stream>>>(x,  xh,  NX);
  mha_cast_f16<<<1024, 256, 0, stream>>>(Wq, wqh, NW);
  mha_cast_f16<<<1024, 256, 0, stream>>>(Wk, wkh, NW);
  mha_cast_f16<<<1024, 256, 0, stream>>>(Wv, wvh, NW);
  mha_cast_f16<<<1024, 256, 0, stream>>>(Wo, woh, NW);

  // 2) Q/K/V projections (block tile 128x128)
  dim3 ggrid(D_MODEL / 128, MROWS / 128);
  mha_gemm_wmma<<<ggrid, 256, 0, stream>>>(xh, wqh, bq, nullptr, qh, D_MODEL, D_MODEL, 1);
  mha_gemm_wmma<<<ggrid, 256, 0, stream>>>(xh, wkh, bk, nullptr, kh, D_MODEL, D_MODEL, 1);
  mha_gemm_wmma<<<ggrid, 256, 0, stream>>>(xh, wvh, bv, nullptr, vt, D_MODEL, D_MODEL, 2);

  // 3) attention
  dim3 agrid(SEQ / 16, NUM_HEADS, BATCH);
  mha_flash_attn<<<agrid, 32, 0, stream>>>(qh, kh, vt, pm, ah);

  // 4) output projection (f32 out)
  mha_gemm_wmma<<<ggrid, 256, 0, stream>>>(ah, woh, bo, out, nullptr, D_MODEL, D_MODEL, 0);
}
